// SyscallLSTM_27479200760318
// MI455X (gfx1250) — compile-verified, matrix operand order checked by reading, not run on hardware
//
#include <hip/hip_runtime.h>

typedef __attribute__((ext_vector_type(16))) _Float16 v16h;
typedef __attribute__((ext_vector_type(8)))  _Float16 v8h;
typedef __attribute__((ext_vector_type(8)))  float    v8f;

#define V_  512
#define D_  512
#define H_  256
#define B_  256
#define T_  512
#define G4  1024  // 4*H

// ---------------------------------------------------------------------------
// Kernel 1: embW[dir][v][c] = b[c] + sum_d emb[v][d] * W[d][c]
// Folds the (B*T)xDx4H input GEMM into a 512x1024 vocabulary table (~1 GFLOP).
// ---------------------------------------------------------------------------
__global__ void embw_kernel(const float* __restrict__ emb,
                            const float* __restrict__ W_f, const float* __restrict__ b_f,
                            const float* __restrict__ W_b, const float* __restrict__ b_b,
                            float* __restrict__ embW) {
  int idx = blockIdx.x * blockDim.x + threadIdx.x;   // 0 .. 2*512*1024-1
  int dir = idx >> 19;
  int r   = idx & ((1 << 19) - 1);
  int v   = r >> 10;
  int c   = r & 1023;
  const float* W = dir ? W_b : W_f;
  const float* b = dir ? b_b : b_f;
  float acc = b[c];
  const float* er = emb + (size_t)v * D_;
  for (int d = 0; d < D_; ++d) acc = fmaf(er[d], W[(size_t)d * G4 + c], acc);
  embW[idx] = acc;
}

// ---------------------------------------------------------------------------
// Kernel 2: pack U (256x1024 f32) into WMMA B-fragment layout, f16.
// Tile (kt,nt) is 32(K) x 16(N); each lane holds 16 halves (32 bytes).
// k_local = e + (e<8?0:8) + (lane<16?0:8); n = nt*16 + (lane&15)
// ---------------------------------------------------------------------------
__global__ void packu_kernel(const float* __restrict__ U_f,
                             const float* __restrict__ U_b,
                             _Float16* __restrict__ Upack) {
  int idx  = blockIdx.x * blockDim.x + threadIdx.x;  // 0 .. 2*8*64*32-1
  int lane = idx & 31;
  int nt   = (idx >> 5) & 63;
  int kt   = (idx >> 11) & 7;
  int dir  = idx >> 14;
  const float* U = dir ? U_b : U_f;
  _Float16* out = Upack + (size_t)idx * 16;
  int n  = nt * 16 + (lane & 15);
  int kb = kt * 32 + ((lane < 16) ? 0 : 8);
  for (int e = 0; e < 16; ++e) {
    int k = kb + e + ((e < 8) ? 0 : 8);
    out[e] = (_Float16)U[(size_t)k * G4 + n];
  }
}

// ---------------------------------------------------------------------------
// Kernel 3: masked LSTM scan. Grid (16 M-tiles, 2 directions), 256 threads.
// Each WG owns 16 batch rows for one direction: the recurrence is fully
// workgroup-local. h kept f16 in LDS; c/h state carried in VGPRs.
// z = embW[token] + h @ U via v_wmma_f32_16x16x32_f16 (f32 accumulate).
// An opaque-per-step scalar offset (inline asm on an SGPR, not the pointer,
// preserving global address-space inference -> global_load_b128, LOADcnt
// only) blocks LICM from hoisting B-fragment loads and spilling to scratch.
// Each fragment's K-reduction is split into two independent WMMA chains.
// ---------------------------------------------------------------------------
__global__ void __launch_bounds__(256) lstm_kernel(const int* __restrict__ tok_ids,
                                                   const float* __restrict__ embW2,
                                                   const _Float16* __restrict__ Upack2,
                                                   float* __restrict__ hout2) {
  __shared__ __align__(32) _Float16 lds_h[16][H_];   // 8 KB

  const int dir   = blockIdx.y;
  const int mtile = blockIdx.x;
  const int tid   = threadIdx.x;
  const int wave  = tid >> 5;
  const int lane  = tid & 31;
  const int l15   = lane & 15;
  const int hi8   = (lane < 16) ? 0 : 8;             // C-frag row offset

  const float* embW = embW2 + (size_t)dir * ((size_t)V_ * G4);
  // Per-lane byte base into Upack; tile (kt,ntg) lives at byte offset
  // (kt*64+ntg)*1024 from here (compile-time immediate offsets, < 2^23).
  const char* upb = (const char*)(Upack2 + (size_t)dir * (8 * 64 * 32 * 16))
                  + (size_t)lane * 32;

  for (int i = tid; i < 16 * H_; i += 256) ((_Float16*)lds_h)[i] = (_Float16)0.f;

  v8f cc[2] = {{}, {}};
  v8f hh[2] = {{}, {}};
  __syncthreads();

  for (int t = 0; t < T_; ++t) {
    const int te = dir ? (T_ - 1 - t) : t;

    // Opaque zero offset, redefined every iteration: the B-fragment loads
    // depend on it, so LICM cannot hoist them out of the time loop (which
    // previously spilled 2KB/lane to scratch). The pointer itself is NOT
    // laundered, so addrspace inference keeps these as global loads.
    size_t tzero = 0;
    asm volatile("" : "+s"(tzero));
    const char* upbase = upb + tzero;

    // A fragments: h(t-1) from LDS, 16x32 f16 tiles (two b128 chunks/lane).
    v16h a[8];
#pragma unroll
    for (int kt = 0; kt < 8; ++kt) {
      const _Float16* hp = &lds_h[l15][kt * 32 + hi8];
      v8h lo = *(const v8h*)hp;
      v8h hi = *(const v8h*)(hp + 16);
      a[kt] = __builtin_shufflevector(lo, hi, 0,1,2,3,4,5,6,7,8,9,10,11,12,13,14,15);
    }

    // Tokens for my 8 rows (broadcast loads, L0-resident).
    int tok[8];
#pragma unroll
    for (int v = 0; v < 8; ++v)
      tok[v] = tok_ids[(size_t)(mtile * 16 + v + hi8) * T_ + te];
    __builtin_amdgcn_sched_barrier(0);

#pragma unroll
    for (int s = 0; s < 2; ++s) {
      // --- GEMM: 4 gate fragments (i,f,g,o) for 16-column sub-tile s ---
      v8f zz[4];
#pragma unroll
      for (int g = 0; g < 4; ++g) {
        const int ntg = g * 16 + wave * 2 + s;
        // Two independent accumulation chains to hide WMMA latency.
        v8f acc0 = {}, acc1 = {};
#pragma unroll
        for (int kt = 0; kt < 4; ++kt) {
          v16h b0 = *(const v16h*)(upbase + (size_t)((2 * kt + 0) * 64 + ntg) * 1024);
          v16h b1 = *(const v16h*)(upbase + (size_t)((2 * kt + 1) * 64 + ntg) * 1024);
          acc0 = __builtin_amdgcn_wmma_f32_16x16x32_f16(
                     false, a[2 * kt + 0], false, b0, (short)0, acc0, false, false);
          acc1 = __builtin_amdgcn_wmma_f32_16x16x32_f16(
                     false, a[2 * kt + 1], false, b1, (short)0, acc1, false, false);
        }
        zz[g] = acc0 + acc1;
        __builtin_amdgcn_sched_barrier(0);
      }

      // --- gates: z += gathered xW, then masked LSTM cell update ---
      const int cbase = wave * 32 + s * 16 + l15;
#pragma unroll
      for (int v = 0; v < 8; ++v) {
        const float* rp = embW + (size_t)tok[v] * G4 + cbase;
        float zi = zz[0][v] + rp[0];
        float zf = zz[1][v] + rp[256];
        float zg = zz[2][v] + rp[512];
        float zo = zz[3][v] + rp[768];
        float ig = 1.f / (1.f + __expf(-zi));
        float fg = 1.f / (1.f + __expf(-zf));
        float gg = 1.f - 2.f / (__expf(2.f * zg) + 1.f);
        float og = 1.f / (1.f + __expf(-zo));
        float cn = fg * cc[s][v] + ig * gg;
        float tc = 1.f - 2.f / (__expf(2.f * cn) + 1.f);
        float hn = og * tc;
        bool  m  = (tok[v] != 0);
        cc[s][v] = m ? cn : cc[s][v];
        hh[s][v] = m ? hn : hh[s][v];
      }
      __builtin_amdgcn_sched_barrier(0);
    }

    __syncthreads();   // all reads of h(t-1) complete before overwriting
#pragma unroll
    for (int s = 0; s < 2; ++s) {
      const int colb = wave * 32 + s * 16 + l15;
#pragma unroll
      for (int v = 0; v < 8; ++v)
        lds_h[v + hi8][colb] = (_Float16)hh[s][v];
    }
    __syncthreads();   // h(t) visible to all waves
  }

  // Final hidden state to workspace (f32).
  float* hout = hout2 + (size_t)dir * (B_ * H_);
#pragma unroll
  for (int s = 0; s < 2; ++s) {
    const int colb = wave * 32 + s * 16 + l15;
    for (int v = 0; v < 8; ++v)
      hout[(size_t)(mtile * 16 + v + hi8) * H_ + colb] = hh[s][v];
  }
}

// ---------------------------------------------------------------------------
// Kernel 4: out[b] = concat(h_fwd[b], h_bwd[b]) @ W_out + b_out   (256x2)
// ---------------------------------------------------------------------------
__global__ void out_kernel(const float* __restrict__ hf, const float* __restrict__ hb,
                           const float* __restrict__ W_out, const float* __restrict__ b_out,
                           float* __restrict__ out) {
  int b = blockIdx.x * blockDim.x + threadIdx.x;
  if (b >= B_) return;
  float a0 = b_out[0], a1 = b_out[1];
  const float* hfr = hf + (size_t)b * H_;
  const float* hbr = hb + (size_t)b * H_;
  for (int k = 0; k < H_; ++k) {
    float v = hfr[k];
    a0 = fmaf(v, W_out[k * 2 + 0], a0);
    a1 = fmaf(v, W_out[k * 2 + 1], a1);
  }
  for (int k = 0; k < H_; ++k) {
    float v = hbr[k];
    a0 = fmaf(v, W_out[(H_ + k) * 2 + 0], a0);
    a1 = fmaf(v, W_out[(H_ + k) * 2 + 1], a1);
  }
  out[b * 2 + 0] = a0;
  out[b * 2 + 1] = a1;
}

// ---------------------------------------------------------------------------
extern "C" void kernel_launch(void* const* d_in, const int* in_sizes, int n_in,
                              void* d_out, int out_size, void* d_ws, size_t ws_size,
                              hipStream_t stream) {
  const int*   tok   = (const int*)  d_in[0];
  const float* emb   = (const float*)d_in[1];
  const float* W_f   = (const float*)d_in[2];
  const float* U_f   = (const float*)d_in[3];
  const float* b_f   = (const float*)d_in[4];
  const float* W_b   = (const float*)d_in[5];
  const float* U_b   = (const float*)d_in[6];
  const float* b_b   = (const float*)d_in[7];
  const float* W_out = (const float*)d_in[8];
  const float* b_out = (const float*)d_in[9];

  char* ws = (char*)d_ws;
  float*    embW  = (float*)ws;                                  // 2*512*1024 f32 = 4 MB
  _Float16* Upack = (_Float16*)(ws + 4ull * 1024 * 1024);        // 2*8*64*32*16 f16 = 1 MB
  float*    hout  = (float*)(ws + 5ull * 1024 * 1024);           // 2*256*256 f32 = 512 KB

  embw_kernel<<<4096, 256, 0, stream>>>(emb, W_f, b_f, W_b, b_b, embW);
  packu_kernel<<<128, 256, 0, stream>>>(U_f, U_b, Upack);
  lstm_kernel<<<dim3(16, 2), 256, 0, stream>>>(tok, embW, Upack, hout);
  out_kernel<<<1, 256, 0, stream>>>(hout, hout + B_ * H_, W_out, b_out, (float*)d_out);
}
